// DetectModel_56126632624643
// MI455X (gfx1250) — matrix operand order repaired and checked by reading, not compiled, not run on hardware
//
#include <hip/hip_runtime.h>

#define N0C 50000
#define N1C 40000
#define N2C 32000
#define EC  600000
#define RC  114
#define MAXT (EC/16 + RC)   // 37614 upper bound on relation tiles

typedef __attribute__((ext_vector_type(16))) _Float16 v16h;
typedef __attribute__((ext_vector_type(8)))  float    v8f;

__device__ __forceinline__ unsigned fenc(float f) {
  unsigned u = __float_as_uint(f);
  return (u & 0x80000000u) ? ~u : (u | 0x80000000u);
}
__device__ __forceinline__ float fdec(unsigned u) {
  unsigned b = (u & 0x80000000u) ? (u ^ 0x80000000u) : ~u;
  return __uint_as_float(b);
}

// ---------------- setup kernels ----------------
__global__ void k_gather_x0(const int* __restrict__ nid, const float* __restrict__ emb,
                            float* __restrict__ x) {
  int i = blockIdx.x * blockDim.x + threadIdx.x;
  if (i >= N0C * 16) return;
  int v = i >> 4, f = i & 15;
  x[i] = emb[nid[v] * 16 + f];
}

__global__ void k_init_edges(const int* __restrict__ ei, int* __restrict__ src,
                             int* __restrict__ dst, int* __restrict__ mask) {
  int e = blockIdx.x * blockDim.x + threadIdx.x;
  if (e >= EC) return;
  src[e] = ei[e]; dst[e] = ei[EC + e]; mask[e] = 1;
}

__global__ void k_hist_type(const int* __restrict__ et, int* __restrict__ cnt) {
  int e = blockIdx.x * blockDim.x + threadIdx.x;
  if (e >= EC) return;
  atomicAdd(&cnt[et[e]], 1);
}

__global__ __launch_bounds__(256) void k_scan_tiles(const int* __restrict__ cnt,
    int* __restrict__ cursor, int* __restrict__ tileRel, int* __restrict__ tileOff,
    int* __restrict__ tileCnt, int* __restrict__ numTiles) {
  __shared__ int c[RC], off[RC + 1], tp[RC + 1];
  int tid = threadIdx.x;
  for (int r = tid; r < RC; r += blockDim.x) c[r] = cnt[r];
  __syncthreads();
  if (tid == 0) {
    int acc = 0, tacc = 0;
    for (int r = 0; r < RC; ++r) {
      off[r] = acc; tp[r] = tacc;
      acc += c[r]; tacc += (c[r] + 15) / 16;
    }
    off[RC] = acc; tp[RC] = tacc; *numTiles = tacc;
  }
  __syncthreads();
  for (int r = tid; r < RC; r += blockDim.x) {
    cursor[r] = off[r];
    int nt = (c[r] + 15) / 16;
    for (int t = 0; t < nt; ++t) {
      int idx = tp[r] + t;
      tileRel[idx] = r;
      tileOff[idx] = off[r] + t * 16;
      int rem = c[r] - t * 16;
      tileCnt[idx] = rem < 16 ? rem : 16;
    }
  }
}

__global__ void k_sort_scatter(const int* __restrict__ et, int* __restrict__ cursor,
                               int* __restrict__ sortedEdge) {
  int e = blockIdx.x * blockDim.x + threadIdx.x;
  if (e >= EC) return;
  int p = atomicAdd(&cursor[et[e]], 1);
  sortedEdge[p] = e;
}

// Wq[r] = W_r @ q, Wk[r] = W_r @ k  (folds the dst-side transform into the logit:
// xi.q = (x W).q = x.(W q), exact by associativity)
__global__ void k_wqk(const float* __restrict__ W, const float* __restrict__ q,
                      const float* __restrict__ k, float* __restrict__ wq,
                      float* __restrict__ wk) {
  int i = blockIdx.x * blockDim.x + threadIdx.x;  // i = r*16 + d
  if (i >= RC * 16) return;
  const float* wrow = W + i * 16;                 // W[r][d][:]
  float aq = 0.0f, ak = 0.0f;
  #pragma unroll
  for (int j = 0; j < 16; ++j) { float w = wrow[j]; aq += w * q[j]; ak += w * k[j]; }
  wq[i] = aq; wk[i] = ak;
}

// ------------- per-edge WMMA transform: xj = x[src] @ W[rel] -------------
// One wave handles 16 edges of one relation. A = src features (16x32, K padded),
// B = W_rel (32x16, rows 16..31 zero). Attention logit computed directly from raw
// features with precomputed Wq/Wk vectors (one shfl_xor to merge feature halves).
__global__ __launch_bounds__(32) void k_edge_transform(
    const float* __restrict__ x, const int* __restrict__ src, const int* __restrict__ dst,
    const float* __restrict__ W, const float* __restrict__ wq, const float* __restrict__ wk,
    const int* __restrict__ sortedEdge, const int* __restrict__ tileRel,
    const int* __restrict__ tileOff, const int* __restrict__ tileCnt,
    const int* __restrict__ numTiles,
    float* __restrict__ alphaOut, float* __restrict__ xjOut) {
  int t = blockIdx.x;
  if (t >= *numTiles) return;
  int lane = threadIdx.x;
  int r    = tileRel[t];
  int base = tileOff[t];
  int cnt  = tileCnt[t];
  int rowm = lane & 15;   // A row (edge within tile) / B column
  int half = lane >> 4;

  // B: lane column n = lane%16; elems 0..7 hold K = 8*half + j, elems 8..15 (K 16..31) = 0
  v16h bm;
  const float* Wr = W + r * 256;
  #pragma unroll
  for (int j = 0; j < 8; ++j) {
    bm[j] = (_Float16)Wr[(half * 8 + j) * 16 + rowm];
    bm[8 + j] = (_Float16)0.0f;
  }

  // lane rowm (both halves) holds edge id for tile row rowm
  int e = sortedEdge[base + (rowm < cnt ? rowm : 0)];
  int d = dst[e], s = src[e];

  const float* wqr = wq + r * 16 + half * 8;
  const float* wkr = wk + r * 16 + half * 8;
  v16h as;
  float aScal = 0.0f;
  #pragma unroll
  for (int j = 0; j < 8; ++j) {
    float xd = x[d * 16 + half * 8 + j];
    float xs = x[s * 16 + half * 8 + j];
    aScal += xd * wqr[j] + xs * wkr[j];
    as[j]     = (_Float16)xs;
    as[8 + j] = (_Float16)0.0f;   // K = 16..31 pad
  }
  v8f c0 = {};
  v8f xj = __builtin_amdgcn_wmma_f32_16x16x32_f16(false, as, false, bm, (short)0, c0, false, false);

  // combine feature halves (feats 0-7 in lanes 0-15, feats 8-15 in lanes 16-31)
  aScal += __shfl_xor(aScal, 16, 32);
  if (half == 0 && rowm < cnt) {
    float a = aScal;
    a = a > 0.0f ? a : 0.2f * a;               // leaky_relu(0.2)
    alphaOut[e] = a;
  }

  // xj stores; D layout: lane n, elem m -> tile row m+8*half, col n
  if (cnt == 16) {
    #pragma unroll
    for (int m = 0; m < 8; ++m) {
      int e2 = __shfl(e, m + half * 8, 32);
      xjOut[e2 * 16 + rowm] = xj[m];
    }
  } else {
    #pragma unroll
    for (int m = 0; m < 8; ++m) {
      int row = m + half * 8;
      int e2 = __shfl(e, row, 32);
      if (row < cnt) xjOut[e2 * 16 + rowm] = xj[m];
    }
  }
}

// ---------------- segment softmax + aggregation ----------------
__global__ void k_seg_max(const float* __restrict__ alpha, const int* __restrict__ dst,
                          const int* __restrict__ mask, unsigned* __restrict__ mEnc) {
  int e = blockIdx.x * blockDim.x + threadIdx.x;
  if (e >= EC || !mask[e]) return;
  atomicMax(&mEnc[dst[e]], fenc(alpha[e]));
}

__global__ void k_seg_exp(const float* __restrict__ alpha, const int* __restrict__ dst,
                          const int* __restrict__ mask, const unsigned* __restrict__ mEnc,
                          float* __restrict__ ex, float* __restrict__ denom) {
  int e = blockIdx.x * blockDim.x + threadIdx.x;
  if (e >= EC) return;
  if (!mask[e]) { ex[e] = 0.0f; return; }
  int d = dst[e];
  float v = expf(alpha[e] - fdec(mEnc[d]));
  ex[e] = v;
  atomicAdd(&denom[d], v);
}

__global__ void k_seg_scatter(const float* __restrict__ ex, const float* __restrict__ denom,
                              const int* __restrict__ dst, const int* __restrict__ mask,
                              const float* __restrict__ xj, float* __restrict__ out) {
  long i = (long)blockIdx.x * blockDim.x + threadIdx.x;
  if (i >= (long)EC * 16) return;
  int e = (int)(i >> 4);
  if (!mask[e]) return;
  int d = dst[e];
  float att = ex[e] / (denom[d] + 1e-16f);
  atomicAdd(&out[d * 16 + (i & 15)], xj[i] * att);
}

// h = relu(out + bias); reduce mean-sum & max per feature (block pre-reduce in LDS)
__global__ __launch_bounds__(256) void k_stats(const float* __restrict__ out,
    const float* __restrict__ bias, int n, float* __restrict__ featSum,
    unsigned* __restrict__ featMax) {
  __shared__ float ssum[16];
  __shared__ unsigned smax[16];
  int tid = threadIdx.x;
  if (tid < 16) { ssum[tid] = 0.0f; smax[tid] = 0u; }
  __syncthreads();
  int i = blockIdx.x * blockDim.x + tid;
  if (i < n * 16) {
    int f = i & 15;
    float h = out[i] + bias[f];
    h = h > 0.0f ? h : 0.0f;
    atomicAdd(&ssum[f], h);
    atomicMax(&smax[f], fenc(h));
  }
  __syncthreads();
  if (tid < 16) {
    atomicAdd(&featSum[tid], ssum[tid]);
    atomicMax(&featMax[tid], smax[tid]);
  }
}

// ---------------- TopK pooling (exact radix-select on tanh scores) ----------------
__global__ void k_score(const float* __restrict__ x, const float* __restrict__ pw, int n,
                        float* __restrict__ score, unsigned* __restrict__ key) {
  int v = blockIdx.x * blockDim.x + threadIdx.x;
  if (v >= n) return;
  float nrm = 0.0f, s = 0.0f;
  #pragma unroll
  for (int f = 0; f < 16; ++f) { float w = pw[f]; nrm += w * w; s += x[v * 16 + f] * w; }
  float sc = tanhf(s / sqrtf(nrm));
  score[v] = sc;
  key[v] = fenc(sc);
}

__global__ void k_hist_hi(const unsigned* __restrict__ key, int n, int* __restrict__ hist) {
  int v = blockIdx.x * blockDim.x + threadIdx.x;
  if (v >= n) return;
  atomicAdd(&hist[key[v] >> 16], 1);
}

__global__ void k_hist_lo(const unsigned* __restrict__ key, int n,
                          const int* __restrict__ sel, int* __restrict__ hist) {
  int v = blockIdx.x * blockDim.x + threadIdx.x;
  if (v >= n) return;
  unsigned kk = key[v];
  if ((int)(kk >> 16) == sel[0]) atomicAdd(&hist[kk & 0xFFFFu], 1);
}

// phase 0: pick hi bin for target k0 -> sel[0]=bin, sel[1]=k remaining
// phase 1: pick lo bin for sel[1]   -> sel[2]=threshKey, sel[3]=tie quota, reset counters
__global__ __launch_bounds__(256) void k_sel_scan(const int* __restrict__ hist, int phase,
                                                  int k0, int* __restrict__ sel) {
  __shared__ int chunk[256];
  int tid = threadIdx.x;
  int sum = 0;
  for (int j = 0; j < 256; ++j) sum += hist[tid * 256 + j];
  chunk[tid] = sum;
  __syncthreads();
  if (tid == 0) {
    int kT = (phase == 0) ? k0 : sel[1];
    int acc = 0, cIdx = 255;
    while (cIdx > 0) {
      if (acc + chunk[cIdx] >= kT) break;
      acc += chunk[cIdx]; --cIdx;
    }
    int b = cIdx * 256 + 255;
    while (b > cIdx * 256) {
      int h = hist[b];
      if (acc + h >= kT) break;
      acc += h; --b;
    }
    if (phase == 0) { sel[0] = b; sel[1] = kT - acc; }
    else {
      sel[2] = (sel[0] << 16) | b;
      sel[3] = kT - acc;     // tie quota at exact threshold key
      sel[4] = 0;            // tie counter
      sel[5] = 0;            // keep counter
    }
  }
}

__global__ void k_assign(const unsigned* __restrict__ key, const float* __restrict__ score,
                         const float* __restrict__ xin, int n, int* __restrict__ sel,
                         int* __restrict__ inv, float* __restrict__ xout) {
  int v = blockIdx.x * blockDim.x + threadIdx.x;
  if (v >= n) return;
  unsigned kk = key[v];
  unsigned th = (unsigned)sel[2];
  bool keep = false;
  if (kk > th) keep = true;
  else if (kk == th) { if (atomicAdd(&sel[4], 1) < sel[3]) keep = true; }
  if (keep) {
    int idx = atomicAdd(&sel[5], 1);
    inv[v] = idx;
    float sc = score[v];
    #pragma unroll
    for (int f = 0; f < 16; ++f) xout[idx * 16 + f] = xin[v * 16 + f] * sc;
  } else {
    inv[v] = -1;
  }
}

__global__ void k_remap(const int* __restrict__ inv, int* __restrict__ src,
                        int* __restrict__ dst, int* __restrict__ mask) {
  int e = blockIdx.x * blockDim.x + threadIdx.x;
  if (e >= EC) return;
  int ns = inv[src[e]], nd = inv[dst[e]];
  int m = (mask[e] && ns >= 0 && nd >= 0) ? 1 : 0;
  src[e] = m ? ns : 0;
  dst[e] = m ? nd : 0;
  mask[e] = m;
}

// ---------------- MLP head ----------------
__global__ __launch_bounds__(128) void k_mlp(const float* __restrict__ featSum,
    const unsigned* __restrict__ featMax,
    const float* __restrict__ l1w, const float* __restrict__ l1b,
    const float* __restrict__ l2w, const float* __restrict__ l2b,
    const float* __restrict__ l3w, const float* __restrict__ l3b,
    float* __restrict__ out) {
  __shared__ float g[96], h1[16], h2[4];
  int tid = threadIdx.x;
  const float ninv[3] = {1.0f / N0C, 1.0f / N1C, 1.0f / N2C};
  if (tid < 96) {
    int layer = tid / 32, w = tid % 32, f = w & 15;
    g[tid] = (w < 16) ? featSum[layer * 16 + f] * ninv[layer]
                      : fdec(featMax[layer * 16 + f]);
  }
  __syncthreads();
  if (tid < 16) {
    float a = l1b[tid];
    for (int i = 0; i < 96; ++i) a += g[i] * l1w[i * 16 + tid];
    h1[tid] = a > 0.0f ? a : 0.0f;
  }
  __syncthreads();
  if (tid < 4) {
    float a = l2b[tid];
    for (int i = 0; i < 16; ++i) a += h1[i] * l2w[i * 4 + tid];
    h2[tid] = a > 0.0f ? a : 0.0f;
  }
  __syncthreads();
  if (tid == 0) {
    float z = l3b[0];
    for (int i = 0; i < 4; ++i) z += h2[i] * l3w[i];
    out[0] = 1.0f / (1.0f + expf(-z));
  }
}

// ---------------- host ----------------
#define GRID(n, b) ((((long)(n)) + (b) - 1) / (b))

extern "C" void kernel_launch(void* const* d_in, const int* in_sizes, int n_in,
                              void* d_out, int out_size, void* d_ws, size_t ws_size,
                              hipStream_t stream) {
  const int*   node_ids   = (const int*)d_in[0];
  const int*   edge_index = (const int*)d_in[1];
  const int*   edge_type  = (const int*)d_in[2];
  const float* emb        = (const float*)d_in[3];
  const float* Wl[3] = {(const float*)d_in[4], (const float*)d_in[8],  (const float*)d_in[12]};
  const float* ql[3] = {(const float*)d_in[5], (const float*)d_in[9],  (const float*)d_in[13]};
  const float* kl[3] = {(const float*)d_in[6], (const float*)d_in[10], (const float*)d_in[14]};
  const float* bl[3] = {(const float*)d_in[7], (const float*)d_in[11], (const float*)d_in[15]};
  const float* pwl[2] = {(const float*)d_in[16], (const float*)d_in[17]};
  const float* l1w = (const float*)d_in[18]; const float* l1b = (const float*)d_in[19];
  const float* l2w = (const float*)d_in[20]; const float* l2b = (const float*)d_in[21];
  const float* l3w = (const float*)d_in[22]; const float* l3b = (const float*)d_in[23];
  float* out = (float*)d_out;

  char* base = (char*)d_ws;
  size_t off = 0;
  auto carve = [&](size_t bytes) -> char* {
    char* p = base + off;
    off = (off + bytes + 255) & ~(size_t)255;
    return p;
  };
  float*    xA         = (float*)carve((size_t)N0C * 16 * 4);
  float*    xB         = (float*)carve((size_t)N0C * 16 * 4);
  int*      sortedEdge = (int*)carve((size_t)EC * 4);
  int*      typeCount  = (int*)carve(RC * 4);
  int*      cursor     = (int*)carve(RC * 4);
  int*      tileRel    = (int*)carve((size_t)MAXT * 4);
  int*      tileOff    = (int*)carve((size_t)MAXT * 4);
  int*      tileCnt    = (int*)carve((size_t)MAXT * 4);
  int*      numTiles   = (int*)carve(256);
  int*      curSrc     = (int*)carve((size_t)EC * 4);
  int*      curDst     = (int*)carve((size_t)EC * 4);
  int*      curMask    = (int*)carve((size_t)EC * 4);
  float*    alphaBuf   = (float*)carve((size_t)EC * 4);
  float*    exBuf      = (float*)carve((size_t)EC * 4);
  float*    xjBuf      = (float*)carve((size_t)EC * 16 * 4);
  unsigned* mEnc       = (unsigned*)carve((size_t)N0C * 4);
  float*    denom      = (float*)carve((size_t)N0C * 4);
  float*    outBuf     = (float*)carve((size_t)N0C * 16 * 4);
  float*    scoreBuf   = (float*)carve((size_t)N0C * 4);
  unsigned* keyBuf     = (unsigned*)carve((size_t)N0C * 4);
  int*      hist       = (int*)carve((size_t)65536 * 4);
  int*      sel        = (int*)carve(256);
  int*      inv        = (int*)carve((size_t)N0C * 4);
  float*    featSum    = (float*)carve(48 * 4);
  unsigned* featMax    = (unsigned*)carve(48 * 4);
  float*    wqBuf      = (float*)carve((size_t)RC * 16 * 4);
  float*    wkBuf      = (float*)carve((size_t)RC * 16 * 4);

  const int nsizes[3] = {N0C, N1C, N2C};

  // one-time per call: edge sort by relation, x0 gather, edge init, pooled-feat reset
  hipMemsetAsync(typeCount, 0, RC * 4, stream);
  k_hist_type<<<GRID(EC, 256), 256, 0, stream>>>(edge_type, typeCount);
  k_scan_tiles<<<1, 256, 0, stream>>>(typeCount, cursor, tileRel, tileOff, tileCnt, numTiles);
  k_sort_scatter<<<GRID(EC, 256), 256, 0, stream>>>(edge_type, cursor, sortedEdge);
  k_gather_x0<<<GRID(N0C * 16, 256), 256, 0, stream>>>(node_ids, emb, xA);
  k_init_edges<<<GRID(EC, 256), 256, 0, stream>>>(edge_index, curSrc, curDst, curMask);
  hipMemsetAsync(featSum, 0, 48 * 4, stream);
  hipMemsetAsync(featMax, 0, 48 * 4, stream);

  float* xcur = xA;
  float* xnext = xB;
  for (int l = 0; l < 3; ++l) {
    int n = nsizes[l];
    hipMemsetAsync(mEnc,  0, (size_t)n * 4, stream);
    hipMemsetAsync(denom, 0, (size_t)n * 4, stream);
    hipMemsetAsync(outBuf, 0, (size_t)n * 16 * 4, stream);

    k_wqk<<<GRID(RC * 16, 256), 256, 0, stream>>>(Wl[l], ql[l], kl[l], wqBuf, wkBuf);
    k_edge_transform<<<MAXT, 32, 0, stream>>>(xcur, curSrc, curDst, Wl[l], wqBuf, wkBuf,
                                              sortedEdge, tileRel, tileOff, tileCnt,
                                              numTiles, alphaBuf, xjBuf);
    k_seg_max<<<GRID(EC, 256), 256, 0, stream>>>(alphaBuf, curDst, curMask, mEnc);
    k_seg_exp<<<GRID(EC, 256), 256, 0, stream>>>(alphaBuf, curDst, curMask, mEnc, exBuf, denom);
    k_seg_scatter<<<GRID((long)EC * 16, 256), 256, 0, stream>>>(exBuf, denom, curDst, curMask,
                                                                xjBuf, outBuf);
    k_stats<<<GRID(n * 16, 256), 256, 0, stream>>>(outBuf, bl[l], n, featSum + l * 16,
                                                   featMax + l * 16);
    if (l < 2) {
      int kkeep = nsizes[l + 1];
      k_score<<<GRID(n, 256), 256, 0, stream>>>(xcur, pwl[l], n, scoreBuf, keyBuf);
      hipMemsetAsync(hist, 0, (size_t)65536 * 4, stream);
      k_hist_hi<<<GRID(n, 256), 256, 0, stream>>>(keyBuf, n, hist);
      k_sel_scan<<<1, 256, 0, stream>>>(hist, 0, kkeep, sel);
      hipMemsetAsync(hist, 0, (size_t)65536 * 4, stream);
      k_hist_lo<<<GRID(n, 256), 256, 0, stream>>>(keyBuf, n, sel, hist);
      k_sel_scan<<<1, 256, 0, stream>>>(hist, 1, kkeep, sel);
      k_assign<<<GRID(n, 256), 256, 0, stream>>>(keyBuf, scoreBuf, xcur, n, sel, inv, xnext);
      k_remap<<<GRID(EC, 256), 256, 0, stream>>>(inv, curSrc, curDst, curMask);
      float* tmp = xcur; xcur = xnext; xnext = tmp;
    }
  }
  k_mlp<<<1, 128, 0, stream>>>(featSum, featMax, l1w, l1b, l2w, l2b, l3w, l3b, out);
}
